// GRU_11553462026467
// MI455X (gfx1250) — matrix operand order repaired and checked by reading, not compiled, not run on hardware
//
#include <hip/hip_runtime.h>

// ---------------------------------------------------------------------------
// GRU on gfx1250 (MI455X): bf16 WMMA 16x16x32, f32 accumulate.
// Fragment layouts follow CDNA5 ISA 7.12.2 exactly:
//   A (16x32 bf16):  lane m = lane&15, hi = lane>>4,
//                    element e=2v+hf -> k = (v>>2)*16 + hi*8 + (v&3)*2 + hf
//   B (32x16 bf16):  lane n = lane&15, hi = lane>>4,
//                    element e       -> k = (e>>3)*16 + hi*8 + (e&7)  (same order as A)
//   C/D (16x16 f32): vgpr v, lane l: m = (l>>4)*8 + v, n = l&15
//
// Step GEMM: each block (8 waves) owns one m-group (64 batch rows). Its whole
// A working set (x_t: 64 KB + h: 128 KB of fragments) is staged once into
// 192 KB of CDNA5 LDS; waves then stream only B (weights) from L2.
// ~65 FLOP per L2 byte vs ~28 without staging.
// ---------------------------------------------------------------------------

typedef __attribute__((ext_vector_type(16))) __bf16 v16bf;
typedef __attribute__((ext_vector_type(8)))  float  v8f;

#define B_  256
#define T_  512
#define I_  512
#define H_  1024
#define O_  512

#define SMEM_X_SH   32768            // 64 frags * 512 shorts  (64 KB)
#define SMEM_H_SH   65536            // 128 frags * 512 shorts (128 KB)
#define SMEM_BYTES  ((SMEM_X_SH + SMEM_H_SH) * 2)   // 196608 B

__device__ __forceinline__ unsigned short f2bf(float f) {
  unsigned u = __builtin_bit_cast(unsigned, f);
  u += 0x7FFFu + ((u >> 16) & 1u);          // round-to-nearest-even
  return (unsigned short)(u >> 16);
}

__device__ __forceinline__ v8f wmma_bf16(v16bf a, v16bf b, v8f c) {
  return __builtin_amdgcn_wmma_f32_16x16x32_bf16(
      /*neg_a=*/false, a, /*neg_b=*/false, b,
      /*c_mod=*/(short)0, c, /*reuse_a=*/false, /*reuse_b=*/false);
}

__device__ __forceinline__ float sigm(float x) {
  return 1.0f / (1.0f + __expf(-x));
}

// 32-byte fragment load from LDS as two 16B halves (only 16B alignment assumed).
__device__ __forceinline__ v16bf ld_frag_lds(const unsigned short* p) {
  struct P { uint4 lo, hi; } t;
  t.lo = *(const uint4*)p;
  t.hi = *(const uint4*)(p + 8);
  return __builtin_bit_cast(v16bf, t);
}

// ---------------------------------------------------------------------------
__global__ void k_zero(float* __restrict__ p, int n) {
  int i = blockIdx.x * 256 + threadIdx.x;
  if (i < n) p[i] = 0.0f;
}

// Weight fp32 [N,K] row-major -> bf16 B-fragment tiles (ntile,ktile), 512 bf16/frag.
__global__ void k_convw(const float* __restrict__ W, unsigned short* __restrict__ frag,
                        int K, int total_lanes) {
  int gid = blockIdx.x * 256 + threadIdx.x;
  if (gid >= total_lanes) return;
  int lane = gid & 31, f = gid >> 5;
  int ktiles = K >> 5;
  int nt = f / ktiles, kt = f - nt * ktiles;
  int n = nt * 16 + (lane & 15);
  int hi = lane >> 4;
  const float* src = W + (size_t)n * K + kt * 32;  // e -> k = (e>>3)*16 + hi*8 + (e&7)
  unsigned short* dst = frag + (size_t)f * 512 + lane * 16;
  #pragma unroll
  for (int e = 0; e < 16; ++e) dst[e] = f2bf(src[(e >> 3) * 16 + hi * 8 + (e & 7)]);
}

// Per-step swizzle: h [B,H] fp32 -> A-fragments; x[:,t,:] fp32 -> A-fragments.
__global__ void k_swz(const float* __restrict__ x, const float* __restrict__ h,
                      unsigned short* __restrict__ xfrag, unsigned short* __restrict__ hfrag,
                      int t) {
  const int HLANES = (B_ / 16) * (H_ / 32) * 32;   // 16384
  const int XLANES = (B_ / 16) * (I_ / 32) * 32;   //  8192
  int gid = blockIdx.x * 256 + threadIdx.x;
  if (gid < HLANES) {
    int lane = gid & 31, f = gid >> 5;
    int mt = f >> 5, kt = f & 31;                  // H/32 = 32 ktiles
    int m = mt * 16 + (lane & 15);
    int hi = lane >> 4;
    const float* src = h + (size_t)m * H_ + kt * 32;
    unsigned short* dst = hfrag + (size_t)f * 512 + lane * 16;
    #pragma unroll
    for (int v = 0; v < 8; ++v) {
      int kb = (v >> 2) * 16 + hi * 8 + (v & 3) * 2;
      dst[2 * v]     = f2bf(src[kb]);
      dst[2 * v + 1] = f2bf(src[kb + 1]);
    }
  } else if (t < T_ && gid < HLANES + XLANES) {
    int g = gid - HLANES;
    int lane = g & 31, f = g >> 5;
    int mt = f >> 4, kt = f & 15;                  // I/32 = 16 ktiles
    int m = mt * 16 + (lane & 15);
    int hi = lane >> 4;
    const float* src = x + (size_t)m * T_ * I_ + (size_t)t * I_ + kt * 32;
    unsigned short* dst = xfrag + (size_t)f * 512 + lane * 16;
    #pragma unroll
    for (int v = 0; v < 8; ++v) {
      int kb = (v >> 2) * 16 + hi * 8 + (v & 3) * 2;
      dst[2 * v]     = f2bf(src[kb]);
      dst[2 * v + 1] = f2bf(src[kb + 1]);
    }
  }
}

// One GRU timestep. Block (8 waves) = one m-group (4 m-tiles, 64 rows) with its
// A-fragments in LDS; wave w handles n-tile (blockIdx&7)*8+w. 4 m-groups x 8
// n-slices = 32 blocks. Each wave: 64x16 output slab, 576 WMMAs.
__global__ void __launch_bounds__(256)
k_step(const unsigned short* __restrict__ xfrag, const unsigned short* __restrict__ hfrag,
       const unsigned short* __restrict__ wxr, const unsigned short* __restrict__ wxc,
       const unsigned short* __restrict__ wxz,
       const unsigned short* __restrict__ whr, const unsigned short* __restrict__ whc,
       const unsigned short* __restrict__ whz,
       const float* __restrict__ bxr, const float* __restrict__ bhr,
       const float* __restrict__ bxc, const float* __restrict__ bhc,
       const float* __restrict__ bxz, const float* __restrict__ bhz,
       float* __restrict__ h) {
  extern __shared__ unsigned short smem[];          // 192 KB dynamic LDS
  unsigned short* xl = smem;                        // 64 x-frags
  unsigned short* hl = smem + SMEM_X_SH;            // 128 h-frags

  const int lane = threadIdx.x & 31;
  const int wid  = threadIdx.x >> 5;
  const int mg   = blockIdx.x >> 3;                 // 0..3  m-group
  const int nt   = (blockIdx.x & 7) * 8 + wid;      // 0..63 n-tile
  const int loff = lane * 16;

  // Stage this m-group's A fragments (contiguous chunks) into LDS.
  {
    const uint4* gx = (const uint4*)(xfrag + (size_t)(mg * 64) * 512);
    uint4* lx = (uint4*)xl;
    #pragma unroll 4
    for (int i = threadIdx.x; i < 4096; i += 256) lx[i] = gx[i];    // 64 KB
    const uint4* gh = (const uint4*)(hfrag + (size_t)(mg * 128) * 512);
    uint4* lh = (uint4*)hl;
    #pragma unroll 4
    for (int i = threadIdx.x; i < 8192; i += 256) lh[i] = gh[i];    // 128 KB
  }
  __syncthreads();

  v8f aR[4], aZ[4], aXC[4], aHC[4];
  #pragma unroll
  for (int i = 0; i < 4; ++i) { aR[i] = (v8f){}; aZ[i] = (v8f){}; aXC[i] = (v8f){}; aHC[i] = (v8f){}; }

  // x_t contribution: K = I_ = 512 -> 16 k-tiles, 3 gates, 4 m-tiles
  #pragma unroll 2
  for (int kt = 0; kt < 16; ++kt) {
    const size_t wb = ((size_t)(nt * 16 + kt)) * 512 + loff;
    __builtin_prefetch(wxr + wb + 2048, 0, 0);      // k-tile +4 ahead
    __builtin_prefetch(wxc + wb + 2048, 0, 0);
    __builtin_prefetch(wxz + wb + 2048, 0, 0);
    v16bf br = *(const v16bf*)(wxr + wb);
    v16bf bc = *(const v16bf*)(wxc + wb);
    v16bf bz = *(const v16bf*)(wxz + wb);
    #pragma unroll
    for (int i = 0; i < 4; ++i) {
      v16bf a = ld_frag_lds(xl + (size_t)(i * 16 + kt) * 512 + loff);
      aR[i]  = wmma_bf16(a, br, aR[i]);
      aXC[i] = wmma_bf16(a, bc, aXC[i]);
      aZ[i]  = wmma_bf16(a, bz, aZ[i]);
    }
  }
  // h contribution: K = H_ = 1024 -> 32 k-tiles, 3 gates, 4 m-tiles
  #pragma unroll 2
  for (int kt = 0; kt < 32; ++kt) {
    const size_t wb = ((size_t)(nt * 32 + kt)) * 512 + loff;
    __builtin_prefetch(whr + wb + 2048, 0, 0);
    __builtin_prefetch(whc + wb + 2048, 0, 0);
    __builtin_prefetch(whz + wb + 2048, 0, 0);
    v16bf br = *(const v16bf*)(whr + wb);
    v16bf bc = *(const v16bf*)(whc + wb);
    v16bf bz = *(const v16bf*)(whz + wb);
    #pragma unroll
    for (int i = 0; i < 4; ++i) {
      v16bf a = ld_frag_lds(hl + (size_t)(i * 32 + kt) * 512 + loff);
      aR[i]  = wmma_bf16(a, br, aR[i]);
      aHC[i] = wmma_bf16(a, bc, aHC[i]);
      aZ[i]  = wmma_bf16(a, bz, aZ[i]);
    }
  }

  // Fused gate math on the WMMA accumulator layout.
  const int n = nt * 16 + (lane & 15);
  const float br_ = bxr[n] + bhr[n];
  const float bz_ = bxz[n] + bhz[n];
  const float bxc_ = bxc[n];
  const float bhc_ = bhc[n];
  #pragma unroll
  for (int i = 0; i < 4; ++i) {
    const int mbase = (mg * 4 + i) * 16 + (lane >> 4) * 8;
    #pragma unroll
    for (int v = 0; v < 8; ++v) {
      float r = sigm(aR[i][v] + br_);
      float c = tanhf(aXC[i][v] + bxc_ + r * (aHC[i][v] + bhc_));
      float z = sigm(aZ[i][v] + bz_);
      float* hp = h + (size_t)(mbase + v) * H_ + n;
      float hold = *hp;
      *hp = (1.0f - z) * hold + z * c;   // same-thread read-modify-write: safe
    }
  }
}

// out = h_final @ Wo^T + bo. Each wave: 64x16 slab; 4 m-groups x 32 n-tiles
// = 128 waves -> 16 blocks x 8 waves.
__global__ void __launch_bounds__(256)
k_out(const unsigned short* __restrict__ hfrag, const unsigned short* __restrict__ wo,
      const float* __restrict__ bo, float* __restrict__ out) {
  const int lane = threadIdx.x & 31;
  const int wave = blockIdx.x * 8 + (threadIdx.x >> 5);   // 0..127
  const int mg = wave >> 5;        // 0..3
  const int nt = wave & 31;        // 0..31 (O/16)
  const int loff = lane * 16;
  v8f acc[4];
  #pragma unroll
  for (int i = 0; i < 4; ++i) acc[i] = (v8f){};
  #pragma unroll 2
  for (int kt = 0; kt < 32; ++kt) {
    v16bf b = *(const v16bf*)(wo + ((size_t)(nt * 32 + kt)) * 512 + loff);
    #pragma unroll
    for (int i = 0; i < 4; ++i) {
      v16bf a = *(const v16bf*)(hfrag + ((size_t)((mg * 4 + i) * 32 + kt)) * 512 + loff);
      acc[i] = wmma_bf16(a, b, acc[i]);
    }
  }
  const int n = nt * 16 + (lane & 15);
  const float bb = bo[n];
  #pragma unroll
  for (int i = 0; i < 4; ++i) {
    const int mbase = (mg * 4 + i) * 16 + (lane >> 4) * 8;
    #pragma unroll
    for (int v = 0; v < 8; ++v)
      out[(size_t)(mbase + v) * O_ + n] = acc[i][v] + bb;
  }
}

// ---------------------------------------------------------------------------
extern "C" void kernel_launch(void* const* d_in, const int* in_sizes, int n_in,
                              void* d_out, int out_size, void* d_ws, size_t ws_size,
                              hipStream_t stream) {
  (void)in_sizes; (void)n_in; (void)out_size; (void)ws_size;
  const float* x   = (const float*)d_in[0];
  const float* Wxr = (const float*)d_in[1];
  const float* bxr = (const float*)d_in[2];
  const float* Whr = (const float*)d_in[3];
  const float* bhr = (const float*)d_in[4];
  const float* Wxc = (const float*)d_in[5];
  const float* bxc = (const float*)d_in[6];
  const float* Whc = (const float*)d_in[7];
  const float* bhc = (const float*)d_in[8];
  const float* Wxz = (const float*)d_in[9];
  const float* bxz = (const float*)d_in[10];
  const float* Whz = (const float*)d_in[11];
  const float* bhz = (const float*)d_in[12];
  const float* Wo  = (const float*)d_in[13];
  const float* bo  = (const float*)d_in[14];

  char* ws = (char*)d_ws;
  const size_t MB = 1048576;
  unsigned short* wxrF = (unsigned short*)(ws + 0 * MB);   // 1 MB each
  unsigned short* wxcF = (unsigned short*)(ws + 1 * MB);
  unsigned short* wxzF = (unsigned short*)(ws + 2 * MB);
  unsigned short* whrF = (unsigned short*)(ws + 3 * MB);   // 2 MB each
  unsigned short* whcF = (unsigned short*)(ws + 5 * MB);
  unsigned short* whzF = (unsigned short*)(ws + 7 * MB);
  unsigned short* woF  = (unsigned short*)(ws + 9 * MB);   // 1 MB
  float*          hf32 = (float*)(ws + 10 * MB);           // 1 MB
  unsigned short* hF   = (unsigned short*)(ws + 11 * MB);  // 512 KB
  unsigned short* xF   = (unsigned short*)(ws + 11 * MB + 524288); // 256 KB

  // Allow 192 KB dynamic LDS for the step kernel (CDNA5 WGP: 320 KB).
  (void)hipFuncSetAttribute((const void*)k_step,
                            hipFuncAttributeMaxDynamicSharedMemorySize, SMEM_BYTES);

  // Weights -> bf16 B-fragments (L2-resident afterwards)
  k_convw<<<128, 256, 0, stream>>>(Wxr, wxrF,  512, 32768);
  k_convw<<<128, 256, 0, stream>>>(Wxc, wxcF,  512, 32768);
  k_convw<<<128, 256, 0, stream>>>(Wxz, wxzF,  512, 32768);
  k_convw<<<256, 256, 0, stream>>>(Whr, whrF, 1024, 65536);
  k_convw<<<256, 256, 0, stream>>>(Whc, whcF, 1024, 65536);
  k_convw<<<256, 256, 0, stream>>>(Whz, whzF, 1024, 65536);
  k_convw<<<128, 256, 0, stream>>>(Wo,  woF,  1024, 32768);

  k_zero<<<(B_ * H_ + 255) / 256, 256, 0, stream>>>(hf32, B_ * H_);

  for (int t = 0; t < T_; ++t) {
    k_swz<<<96, 256, 0, stream>>>(x, hf32, xF, hF, t);
    k_step<<<32, 256, SMEM_BYTES, stream>>>(xF, hF, wxrF, wxcF, wxzF, whrF, whcF, whzF,
                                            bxr, bhr, bxc, bhc, bxz, bhz, hf32);
  }
  k_swz<<<96, 256, 0, stream>>>(x, hf32, xF, hF, T_);   // final h fragments only
  k_out<<<16, 256, 0, stream>>>(hF, woF, bo, (float*)d_out);
}
